// ContextCorrelationEncoder_55929064129156
// MI455X (gfx1250) — compile-verified, matrix operand order checked by reading, not run on hardware
//
#include <hip/hip_runtime.h>
#include <hip/hip_bf16.h>

#define HD __device__ __forceinline__

typedef __attribute__((ext_vector_type(16))) unsigned short v16u;
typedef __attribute__((ext_vector_type(16))) __bf16        v16bf;
typedef __attribute__((ext_vector_type(8)))  float         v8f;
typedef __attribute__((ext_vector_type(4)))  unsigned int  v4u;
typedef __attribute__((ext_vector_type(8)))  int           v8i;
typedef __attribute__((ext_vector_type(4)))  int           v4i;

#define BB 4
#define CC 64
#define HH 192
#define WW 192
#define HWSZ (HH * WW)          // 36864
#define NPIX (BB * HWSZ)        // 147456
#define KCONV 576               // 9 taps * 64 cin
#define KQ 160                  // 145 padded to 160

// ---------- bf16 helpers (manual RNE conversion; storage as u16) ----------
HD unsigned short f2bf(float f) {
    unsigned int u = __float_as_uint(f);
    u += 0x7FFFu + ((u >> 16) & 1u);
    return (unsigned short)(u >> 16);
}
HD float bf2f(unsigned short h) {
    return __uint_as_float(((unsigned int)h) << 16);
}

// ---------- WMMA wrapper: D = A(bf16 16x32) * B(bf16 32x16) + C(f32 16x16) ----------
HD v8f wmma_bf16(v16u a, v16u b, v8f c) {
    return __builtin_amdgcn_wmma_f32_16x16x32_bf16(
        /*neg_a=*/false, __builtin_bit_cast(v16bf, a),
        /*neg_b=*/false, __builtin_bit_cast(v16bf, b),
        /*c_mod=*/(short)0, c,
        /*reuse_a=*/false, /*reuse_b=*/false);
}

// ---------- TDM: stage a contiguous run of bf16 weights into LDS ----------
// Builds a D# (group0/group1 per CDNA5 ISA):
//   count=1, lds_addr, global_addr=gsrc, type=2("image"),
//   data_size=2B, 1 row of nElems elements (tile_dim0=tensor_dim0=nElems).
// Issued by one wave; completion via TENSORcnt.
#if __has_builtin(__builtin_amdgcn_tensor_load_to_lds)
#define USE_TDM 1
HD void tdm_load_to_lds(const unsigned short* gsrc, unsigned int ldsByteOff,
                        unsigned int nElems) {
    unsigned long long ga = (unsigned long long)gsrc;
    v4u g0 = {};
    g0.x = 1u;                                            // count=1 (valid), user mode
    g0.y = ldsByteOff;                                    // lds_addr
    g0.z = (unsigned int)(ga & 0xFFFFFFFFu);              // global_addr[31:0]
    g0.w = (unsigned int)((ga >> 32) & 0x01FFFFFFu)       // global_addr[56:32]
         | (2u << 30);                                    // type=2
    v8i g1 = {};
    g1[0] = (1 << 16);                                    // data_size=1 -> 2 bytes; wg_mask=0
    g1[1] = (int)((nElems & 0xFFFFu) << 16);              // tensor_dim0[15:0]  (bits 63:48)
    g1[2] = (int)((nElems >> 16) & 0xFFFFu)               // tensor_dim0[31:16] (bits 79:64)
          | (1 << 16);                                    // tensor_dim1=1      (bits 111:80)
    g1[3] = (int)((nElems & 0xFFFFu) << 16);              // tile_dim0          (bits 127:112)
    g1[4] = 0;                                            // tile_dim1=0 (unused), tile_dim2=0
    g1[5] = (int)nElems;                                  // tensor_dim0_stride (bits 207:160)
    v4i z4 = {};
#if defined(__clang_major__) && (__clang_major__ >= 23)
    v8i z8 = {};
    __builtin_amdgcn_tensor_load_to_lds(g0, g1, z4, z4, z8, 0);
#else
    __builtin_amdgcn_tensor_load_to_lds(g0, g1, z4, z4, 0);
#endif
    __builtin_amdgcn_s_wait_tensorcnt(0);
}
#else
#define USE_TDM 0
#endif

// Stage weights into LDS: TDM when available, cooperative copy otherwise.
// The TDM write to LDS is invisible to LLVM alias analysis (the builtin takes
// descriptors, not the LDS pointer), so after the barrier we make the LDS
// pointer escape into an asm with a memory clobber — otherwise the compiler
// proves the shared array is never stored and folds all reads to undef
// (which is exactly what collapsed the WMMA chains in the previous round).
HD void stage_weights(unsigned short* lds, const unsigned short* gsrc, int nElems,
                      int tid, int nthreads) {
#if USE_TDM
    if (tid < 32)
        tdm_load_to_lds(gsrc, (unsigned int)(size_t)lds, (unsigned int)nElems);
    __syncthreads();
    asm volatile("" : : "s"(lds) : "memory");   // TDM wrote LDS behind AA's back
#else
    for (int i = tid * 8; i < nElems; i += nthreads * 8)
        *(v4u*)(lds + i) = *(const v4u*)(gsrc + i);
    __syncthreads();
#endif
}

// =====================================================================
// Prep: NCHW f32 -> channels-last bf16  ([B][H][W][64])
// =====================================================================
__global__ void prep_input(const float* __restrict__ src,
                           unsigned short* __restrict__ dst, int total) {
    int i = blockIdx.x * blockDim.x + threadIdx.x;
    if (i >= total) return;
    int c   = i & 63;
    int pg  = i >> 6;               // b*H*W + hw
    int b   = pg / HWSZ;
    int hw  = pg % HWSZ;
    dst[i] = f2bf(src[((size_t)b * CC + c) * HWSZ + hw]);
}

// Prep: 3x3 weights OIHW -> implicit-GEMM A matrix [cout][tap*64+cin] bf16
__global__ void prep_w3(const float* __restrict__ wk, const float* __restrict__ wq,
                        unsigned short* __restrict__ wkA, unsigned short* __restrict__ wqA) {
    int i = blockIdx.x * blockDim.x + threadIdx.x;
    if (i >= 2 * CC * KCONV) return;
    int sel  = i / (CC * KCONV);
    int r    = i % (CC * KCONV);
    int cout = r / KCONV;
    int kk   = r % KCONV;
    int tap  = kk >> 6;
    int cin  = kk & 63;
    const float* src = sel ? wq : wk;
    unsigned short* dst = sel ? wqA : wkA;
    dst[r] = f2bf(src[((size_t)cout * CC + cin) * 9 + tap]);
}

// Prep: 1x1 weights [64][145] -> [64][160] bf16 (zero-padded K)
__global__ void prep_q(const float* __restrict__ qw, unsigned short* __restrict__ qA) {
    int i = blockIdx.x * blockDim.x + threadIdx.x;
    if (i >= 64 * KQ) return;
    int cout = i / KQ;
    int kk   = i % KQ;
    qA[i] = (kk < 145) ? f2bf(qw[(size_t)cout * 145 + kk]) : (unsigned short)0;
}

// Prep: fold conv bias + BN into per-channel scale/bias (3 groups of 64)
__global__ void prep_bn(
    const float* wkb, const float* wkg, const float* wkbe, const float* wkm, const float* wkv,
    const float* wqb, const float* wqg, const float* wqbe, const float* wqm, const float* wqv,
    const float* qb,  const float* qg,  const float* qbe,  const float* qm,  const float* qv,
    float* __restrict__ bnS, float* __restrict__ bnB) {
    int i = threadIdx.x;
    if (i >= 192) return;
    int grp = i >> 6, c = i & 63;
    const float *cb, *g, *be, *m, *v;
    if (grp == 0)      { cb = wkb; g = wkg; be = wkbe; m = wkm; v = wkv; }
    else if (grp == 1) { cb = wqb; g = wqg; be = wqbe; m = wqm; v = wqv; }
    else               { cb = qb;  g = qg;  be = qbe;  m = qm;  v = qv;  }
    float inv = g[c] * rsqrtf(v[c] + 1e-5f);
    bnS[i] = inv;
    bnB[i] = (cb[c] - m[c]) * inv + be[c];
}

// =====================================================================
// 3x3 conv + BN + ReLU as implicit GEMM.
// Weights (64x576 bf16 = 72KB, reused by every pixel) are TDM-staged into
// LDS once per workgroup; A-fragments come from ds_load, B-fragments are
// contiguous 32B global loads of channels-last input (served by WGP$/L2).
// Wave tile: 64 cout x 16 pixels; 18 K-steps x 4 WMMAs = 72 v_wmma/wave.
// =====================================================================
__global__ void __launch_bounds__(128)
conv3x3_bn_relu_wmma(const unsigned short* __restrict__ in,   // [B][H][W][64] bf16
                     const unsigned short* __restrict__ wA,   // [64][576] bf16
                     const float* __restrict__ bnS, const float* __restrict__ bnB,
                     unsigned short* __restrict__ outBuf, int outStride, int outOffset,
                     int zeroPad) {
    __shared__ __attribute__((aligned(32))) unsigned short wlds[CC * KCONV]; // 72 KB

    const int lane  = threadIdx.x & 31;
    const int wave  = threadIdx.x >> 5;
    const int n     = lane & 15;       // pixel within tile (N index)
    const int khalf = lane >> 4;       // K half-selector per WMMA layout

    stage_weights(wlds, wA, CC * KCONV, threadIdx.x, blockDim.x);

    const int tilesPerRow = WW / 64;   // 3
    int blk = blockIdx.x;
    int b   = blk / (HH * tilesPerRow);
    int rem = blk % (HH * tilesPerRow);
    int h   = rem / tilesPerRow;
    int wt  = rem % tilesPerRow;
    int w0  = wt * 64 + wave * 16;

    v8f acc[4] = {};

#pragma unroll
    for (int tap = 0; tap < 9; ++tap) {
        int dy = tap / 3 - 1, dx = tap % 3 - 1;
        int hh = h + dy;
        int ww = w0 + n + dx;
        bool inb = (hh >= 0) && (hh < HH) && (ww >= 0) && (ww < WW);
        const unsigned short* bp =
            in + (((size_t)b * HH + (inb ? hh : 0)) * WW + (inb ? ww : 0)) * 64;
#pragma unroll
        for (int kh = 0; kh < 2; ++kh) {
            int ksub = kh * 32 + khalf * 16;
            v16u bu = {};
            if (inb) bu = *(const v16u*)(bp + ksub);
            const unsigned short* ap = wlds + (tap * 64 + ksub) + n * KCONV;
            v16u a0 = *(const v16u*)(ap);
            v16u a1 = *(const v16u*)(ap + 16 * KCONV);
            v16u a2 = *(const v16u*)(ap + 32 * KCONV);
            v16u a3 = *(const v16u*)(ap + 48 * KCONV);
            acc[0] = wmma_bf16(a0, bu, acc[0]);
            acc[1] = wmma_bf16(a1, bu, acc[1]);
            acc[2] = wmma_bf16(a2, bu, acc[2]);
            acc[3] = wmma_bf16(a3, bu, acc[3]);
        }
    }

    size_t pix = ((size_t)b * HH + h) * WW + (w0 + n);
    unsigned short* op = outBuf + pix * (size_t)outStride + outOffset;
#pragma unroll
    for (int t = 0; t < 4; ++t) {
#pragma unroll
        for (int j = 0; j < 8; ++j) {
            int cout = t * 16 + khalf * 8 + j;      // C-matrix layout: VGPR j, lane half
            float v  = fmaf(acc[t][j], bnS[cout], bnB[cout]);
            op[cout] = f2bf(fmaxf(v, 0.0f));
        }
    }
    if (zeroPad && lane < 16) {                      // zero x[pix][145..159]
        unsigned short* pp = outBuf + pix * (size_t)outStride;
#pragma unroll
        for (int k = 145; k < 160; ++k) pp[k] = 0;
    }
}

// =====================================================================
// 81-shift local correlation: per-pixel batched dot over 64 channels.
// Not GEMM-shaped -> VALU FMA. f1 vector staged in LDS once per block.
// Writes corr into x[pix][0..80]; OOB shifts are exactly 0 (zero pad).
// =====================================================================
__global__ void __launch_bounds__(96)
corr_kernel(const unsigned short* __restrict__ f2buf,   // [B][H][W][64] bf16
            unsigned short* __restrict__ xbuf) {        // [pix][160] bf16
    __shared__ float f1s[64];
    int pix = blockIdx.x;
    int b   = pix / HWSZ;
    int hw  = pix % HWSZ;
    int h   = hw / WW, w = hw % WW;
    int t   = threadIdx.x;
    unsigned short* xp = xbuf + (size_t)pix * KQ;
    if (t < 64) f1s[t] = bf2f(xp[81 + t]);
    __syncthreads();
    if (t < 81) {
        int a = t / 9, bb = t % 9;                   // channel order = a*9+bb
        int hh = h + bb - 4, ww = w + a - 4;
        float acc = 0.0f;
        if (hh >= 0 && hh < HH && ww >= 0 && ww < WW) {
            const unsigned short* p = f2buf + (((size_t)b * HH + hh) * WW + ww) * 64;
#pragma unroll 8
            for (int c = 0; c < 64; ++c) acc = fmaf(f1s[c], bf2f(p[c]), acc);
        }
        xp[t] = f2bf(acc * 0.125f);                  // 1/sqrt(64)
    }
}

// =====================================================================
// 1x1 conv (K=145 padded to 160) + BN + ReLU as a straight GEMM.
// 20KB weight matrix TDM-staged to LDS once per workgroup.
// Wave tile: 64 cout x 16 pixels; 5 K-steps x 4 WMMAs. fp32 NCHW out.
// =====================================================================
__global__ void __launch_bounds__(128)
conv1x1_bn_relu_wmma(const unsigned short* __restrict__ xbuf,  // [pix][160] bf16
                     const unsigned short* __restrict__ qA,    // [64][160] bf16
                     const float* __restrict__ bnS, const float* __restrict__ bnB,
                     float* __restrict__ out) {
    __shared__ __attribute__((aligned(32))) unsigned short qlds[CC * KQ];   // 20 KB

    const int lane  = threadIdx.x & 31;
    const int wave  = threadIdx.x >> 5;
    const int n     = lane & 15;
    const int khalf = lane >> 4;
    size_t pix = (size_t)blockIdx.x * 64 + wave * 16 + n;

    stage_weights(qlds, qA, CC * KQ, threadIdx.x, blockDim.x);

    v8f acc[4] = {};
#pragma unroll
    for (int k = 0; k < 5; ++k) {
        int ksub = k * 32 + khalf * 16;
        v16u bu = *(const v16u*)(xbuf + pix * KQ + ksub);
        const unsigned short* ap = qlds + ksub + n * KQ;
        v16u a0 = *(const v16u*)(ap);
        v16u a1 = *(const v16u*)(ap + 16 * KQ);
        v16u a2 = *(const v16u*)(ap + 32 * KQ);
        v16u a3 = *(const v16u*)(ap + 48 * KQ);
        acc[0] = wmma_bf16(a0, bu, acc[0]);
        acc[1] = wmma_bf16(a1, bu, acc[1]);
        acc[2] = wmma_bf16(a2, bu, acc[2]);
        acc[3] = wmma_bf16(a3, bu, acc[3]);
    }

    int bI = (int)(pix / HWSZ);
    int hw = (int)(pix % HWSZ);
#pragma unroll
    for (int t = 0; t < 4; ++t) {
#pragma unroll
        for (int j = 0; j < 8; ++j) {
            int cout = t * 16 + khalf * 8 + j;
            float v  = fmaf(acc[t][j], bnS[cout], bnB[cout]);
            out[((size_t)bI * 64 + cout) * HWSZ + hw] = fmaxf(v, 0.0f);
        }
    }
}

// =====================================================================
extern "C" void kernel_launch(void* const* d_in, const int* in_sizes, int n_in,
                              void* d_out, int out_size, void* d_ws, size_t ws_size,
                              hipStream_t stream) {
    const float* fm1    = (const float*)d_in[0];
    const float* fm2    = (const float*)d_in[1];
    const float* wk_w   = (const float*)d_in[2];
    const float* wk_b   = (const float*)d_in[3];
    const float* wk_g   = (const float*)d_in[4];
    const float* wk_be  = (const float*)d_in[5];
    const float* wk_m   = (const float*)d_in[6];
    const float* wk_v   = (const float*)d_in[7];
    const float* wq_w   = (const float*)d_in[8];
    const float* wq_b   = (const float*)d_in[9];
    const float* wq_g   = (const float*)d_in[10];
    const float* wq_be  = (const float*)d_in[11];
    const float* wq_m   = (const float*)d_in[12];
    const float* wq_v   = (const float*)d_in[13];
    const float* q_w    = (const float*)d_in[14];
    const float* q_b    = (const float*)d_in[15];
    const float* q_g    = (const float*)d_in[16];
    const float* q_be   = (const float*)d_in[17];
    const float* q_m    = (const float*)d_in[18];
    const float* q_v    = (const float*)d_in[19];

    char* ws = (char*)d_ws;
    const size_t o_in1 = 0;
    const size_t o_in2 = o_in1 + (size_t)NPIX * 64 * 2;   //  18,874,368
    const size_t o_x   = o_in2 + (size_t)NPIX * 64 * 2;   //  37,748,736
    const size_t o_f2  = o_x   + (size_t)NPIX * KQ * 2;   //  84,934,656
    const size_t o_wk  = o_f2  + (size_t)NPIX * 64 * 2;   // 103,809,024
    const size_t o_wq  = o_wk  + (size_t)CC * KCONV * 2;
    const size_t o_qw  = o_wq  + (size_t)CC * KCONV * 2;
    const size_t o_bn  = o_qw  + (size_t)64 * KQ * 2;

    unsigned short* in1 = (unsigned short*)(ws + o_in1);
    unsigned short* in2 = (unsigned short*)(ws + o_in2);
    unsigned short* xb  = (unsigned short*)(ws + o_x);
    unsigned short* f2b = (unsigned short*)(ws + o_f2);
    unsigned short* wkA = (unsigned short*)(ws + o_wk);
    unsigned short* wqA = (unsigned short*)(ws + o_wq);
    unsigned short* qA  = (unsigned short*)(ws + o_qw);
    float* bnS = (float*)(ws + o_bn);
    float* bnB = bnS + 192;

    // ---- prep ----
    const int totalIn = NPIX * 64;
    prep_input<<<(totalIn + 255) / 256, 256, 0, stream>>>(fm1, in1, totalIn);
    prep_input<<<(totalIn + 255) / 256, 256, 0, stream>>>(fm2, in2, totalIn);
    prep_w3<<<(2 * CC * KCONV + 255) / 256, 256, 0, stream>>>(wk_w, wq_w, wkA, wqA);
    prep_q<<<(64 * KQ + 255) / 256, 256, 0, stream>>>(q_w, qA);
    prep_bn<<<1, 192, 0, stream>>>(wk_b, wk_g, wk_be, wk_m, wk_v,
                                   wq_b, wq_g, wq_be, wq_m, wq_v,
                                   q_b,  q_g,  q_be,  q_m,  q_v,
                                   bnS, bnB);

    // ---- two 3x3 convs (implicit GEMM, WMMA bf16, TDM-staged weights) ----
    dim3 g3(BB * HH * (WW / 64));
    conv3x3_bn_relu_wmma<<<g3, 128, 0, stream>>>(in1, wkA, bnS + 0,  bnB + 0,
                                                 xb,  KQ, 81, 1);   // f1 -> x[81..144], pad 145..159
    conv3x3_bn_relu_wmma<<<g3, 128, 0, stream>>>(in2, wqA, bnS + 64, bnB + 64,
                                                 f2b, 64, 0, 0);    // f2 -> own buffer

    // ---- 81-shift correlation -> x[0..80] ----
    corr_kernel<<<NPIX, 96, 0, stream>>>(f2b, xb);

    // ---- 1x1 conv GEMM + BN + ReLU -> fp32 NCHW out ----
    conv1x1_bn_relu_wmma<<<NPIX / 64, 128, 0, stream>>>(xb, qA, bnS + 128, bnB + 128,
                                                        (float*)d_out);
}
